// DeepSeekRouter_68272800137431
// MI455X (gfx1250) — matrix operand order repaired and checked by reading, not compiled, not run on hardware
//
#include <hip/hip_runtime.h>
#include <hip/hip_bf16.h>

// ---------------------------------------------------------------------------
// DeepSeek-style MoE router for MI455X (gfx1250, wave32, WMMA).
// Memory-bound (256MB hidden f32 stream @ 23.3TB/s ~= 11.5us floor).
// GEMM: v_wmma_f32_16x16x32_bf16. Gate tiles: GLOBAL_LOAD_ASYNC_TO_LDS_B128
// (ASYNCcnt-tracked copy engine). Double-buffered LDS, 1 barrier per chunk.
// ---------------------------------------------------------------------------

#define TOPK   6
#define NEXP   64
#define HDIM   4096
#define NTOK   16384
#define KT     64          // K-chunk per LDS stage
#define RP     72          // LDS row pitch in bf16 halves (144B, bank-skewed)

#define HID_B  (128 * RP * 2)          // 18432 B hidden tile
#define GATE_B (NEXP * RP * 2)         //  9216 B gate tile
#define BUF_B  (HID_B + GATE_B)        // 27648 B per buffer

typedef __attribute__((ext_vector_type(16))) __bf16 bf16x16;
typedef __attribute__((ext_vector_type(8)))  __bf16 bf16x8;
typedef __attribute__((ext_vector_type(4)))  __bf16 bf16x4;
typedef __attribute__((ext_vector_type(8)))  float  f32x8;

union FragU { bf16x16 v; bf16x8 h[2]; };

#define WS_GATE_BYTES ((size_t)NEXP * HDIM * sizeof(__bf16))   // 512 KB

// Output layout (flat f32 concat, reference return order)
#define OUT_RW    0          // routing_weights [NTOK,6]
#define OUT_SEL   98304      // selected_experts [NTOK,6] (float(idx))
#define OUT_PROBS 196608     // probs [NTOK,64]
#define OUT_AUX   1245184    // aux_loss scalar

// ---------------------------------------------------------------------------
// Kernel 0: gate_w f32 -> bf16 into ws; zero global accumulators each launch.
// ---------------------------------------------------------------------------
__global__ void router_prep(const float* __restrict__ gate_w, void* __restrict__ ws) {
  __bf16* gbf = (__bf16*)ws;
  int i0 = (blockIdx.x * blockDim.x + threadIdx.x) * 8;
  #pragma unroll
  for (int j = 0; j < 8; ++j) gbf[i0 + j] = (__bf16)gate_w[i0 + j];
  if (blockIdx.x == 0 && threadIdx.x < 129) {
    float* acc = (float*)((char*)ws + WS_GATE_BYTES);
    acc[threadIdx.x] = 0.0f;
  }
}

// ---------------------------------------------------------------------------
// Kernel 1: fused GEMM (WMMA) + softmax + top-6 + aux partial reductions.
// 128 blocks x 256 threads (8 wave32); each wave: 16 tokens x 64 experts.
// ---------------------------------------------------------------------------
__global__ void __launch_bounds__(256)
router_main(const float* __restrict__ hidden,
            const float* __restrict__ bias,
            const __bf16* __restrict__ gate_bf,
            float* __restrict__ out,
            float* __restrict__ gacc) {
  // Double-buffered staging; epilogue logits scratch aliases buffer 0/1.
  __shared__ __align__(16) unsigned char smem[2 * BUF_B];   // 55296 B
  __shared__ float accLds[129];                             // cnt[64], psum[64], zsum

  const int  tid    = threadIdx.x;
  const int  wv     = tid >> 5;
  const int  lane   = tid & 31;
  const int  l16    = lane & 15;
  const bool hiHalf = lane >= 16;
  const int  tok0   = blockIdx.x * 128;

  if (tid < 129) accLds[tid] = 0.0f;

  f32x8 acc[4];
  #pragma unroll
  for (int nt = 0; nt < 4; ++nt)
    #pragma unroll
    for (int r = 0; r < 8; ++r) acc[nt][r] = 0.0f;

  // --- staging helpers (issue only; completion enforced at loop top) ---
  auto stageHidden = [&](int kc, int bufi) {
    __bf16* hl = (__bf16*)(smem + bufi * BUF_B);
    #pragma unroll
    for (int j = 0; j < 8; ++j) {
      int idx = j * 256 + tid;          // 0..2047
      int t   = idx >> 4;               // token row 0..127
      int q   = idx & 15;               // float4 slot within 64 floats
      const float* gp = hidden + (size_t)(tok0 + t) * HDIM + kc + q * 4;
      float4 v = *(const float4*)gp;
      if (kc + 2 * KT < HDIM) __builtin_prefetch(gp + 2 * KT, 0, 1);  // global_prefetch_b8
      bf16x4 b;
      b.x = (__bf16)v.x; b.y = (__bf16)v.y; b.z = (__bf16)v.z; b.w = (__bf16)v.w;
      *(bf16x4*)&hl[t * RP + q * 4] = b;
    }
  };
  auto stageGateAsync = [&](int kc, int bufi) {
    __bf16* gl = (__bf16*)(smem + bufi * BUF_B + HID_B);
    #pragma unroll
    for (int j = 0; j < 2; ++j) {
      int idx = j * 256 + tid;          // 0..511 : 16B chunks
      int r   = idx >> 3;               // expert row 0..63
      int s   = idx & 7;                // 16B segment within 128B row
      unsigned ldsa = (unsigned)(size_t)&gl[r * RP + s * 8];
      unsigned long long ga =
          (unsigned long long)(size_t)(gate_bf + (size_t)r * HDIM + kc + s * 8);
      // CDNA5 async copy engine: memory -> LDS, tracked by ASYNCcnt.
      asm volatile("global_load_async_to_lds_b128 %0, %1, off"
                   :: "v"(ldsa), "v"(ga) : "memory");
    }
  };

  const int nIter = HDIM / KT;          // 64
  stageHidden(0, 0);
  stageGateAsync(0, 0);

  for (int i = 0; i < nIter; ++i) {
    const int kc = i * KT;
    const int cb = i & 1;
    // Only the current buffer's async loads are outstanding here.
    asm volatile("s_wait_asynccnt 0" ::: "memory");
    __syncthreads();
    if (i + 1 < nIter) {                // issue next chunk; overlaps with WMMA
      stageHidden(kc + KT, cb ^ 1);
      stageGateAsync(kc + KT, cb ^ 1);
    }

    const __bf16* hl = (const __bf16*)(smem + cb * BUF_B);
    const __bf16* gl = (const __bf16*)(smem + cb * BUF_B + HID_B);
    const int arow = wv * 16 + l16;     // A lane holds token row M = l16 (ISA 7.12.2)
    #pragma unroll
    for (int ks = 0; ks < KT; ks += 32) {
      FragU a;
      int ao = arow * RP + ks + (hiHalf ? 8 : 0);
      a.h[0] = *(const bf16x8*)&hl[ao];
      a.h[1] = *(const bf16x8*)&hl[ao + 16];
      #pragma unroll
      for (int nt = 0; nt < 4; ++nt) {
        FragU b;
        int bo = (nt * 16 + l16) * RP + ks + (hiHalf ? 8 : 0);
        b.h[0] = *(const bf16x8*)&gl[bo];
        b.h[1] = *(const bf16x8*)&gl[bo + 16];
        acc[nt] = __builtin_amdgcn_wmma_f32_16x16x32_bf16(
            false, a.v, false, b.v, (short)0, acc[nt], false, false);
      }
    }
  }

  __syncthreads();                      // all WMMA reads done; smem now reusable
  float* logLds = (float*)smem;         // [8 waves][16 tok][65] = 33280 B

  // ---- spill logits (+bias) to per-wave LDS scratch (intra-wave only) ----
  #pragma unroll
  for (int nt = 0; nt < 4; ++nt) {
    int e = nt * 16 + l16;
    float bvv = bias[e];
    #pragma unroll
    for (int r = 0; r < 8; ++r) {
      int m = r + (hiHalf ? 8 : 0);
      logLds[(wv * 16 + m) * 65 + e] = acc[nt][r] + bvv;
    }
  }

  // ---- per-token epilogue: lanes 0..15 each own one of the wave's 16 tokens ----
  if (!hiHalf) {
    float* row = &logLds[(wv * 16 + l16) * 65];
    int tok = tok0 + wv * 16 + l16;
    float mx = -3.0e38f;
    for (int e = 0; e < NEXP; ++e) mx = fmaxf(mx, row[e]);
    float s = 0.0f;
    for (int e = 0; e < NEXP; ++e) { float v = __expf(row[e] - mx); row[e] = v; s += v; }
    float rs  = 1.0f / s;
    float lse = mx + __logf(s);

    float* probs = out + OUT_PROBS + (size_t)tok * NEXP;
    for (int e = 0; e < NEXP; ++e) {
      float pr = row[e] * rs;
      probs[e] = pr;
      atomicAdd(&accLds[64 + e], pr);           // per-expert prob sum
    }

    unsigned long long msk = 0;
    int   sel[TOPK];
    float sv[TOPK];
    float tw = 0.0f;
    #pragma unroll
    for (int k = 0; k < TOPK; ++k) {
      float bvv = -1.0f; int bi = 0;
      for (int e = 0; e < NEXP; ++e) {
        float v = row[e];
        if (!((msk >> e) & 1ull) && v > bvv) { bvv = v; bi = e; }
      }
      msk |= 1ull << bi;
      sel[k] = bi;
      sv[k]  = bvv * rs;
      tw    += sv[k];
      atomicAdd(&accLds[bi], 1.0f);             // tokens-per-expert count
    }
    float itw = 1.0f / tw;
    #pragma unroll
    for (int k = 0; k < TOPK; ++k) {
      out[OUT_RW  + (size_t)tok * TOPK + k] = sv[k] * itw;
      out[OUT_SEL + (size_t)tok * TOPK + k] = (float)sel[k];
    }
    atomicAdd(&accLds[128], lse * lse);
  }

  __syncthreads();
  if (tid < 129) atomicAdd(&gacc[tid], accLds[tid]);
}

// ---------------------------------------------------------------------------
// Kernel 2: finalize aux loss.
// ---------------------------------------------------------------------------
__global__ void router_fin(const float* __restrict__ gacc, float* __restrict__ out) {
  if (threadIdx.x == 0 && blockIdx.x == 0) {
    float tot = 0.0f;
    for (int e = 0; e < NEXP; ++e) tot += gacc[e];
    float lb = 0.0f;
    for (int e = 0; e < NEXP; ++e)
      lb += (gacc[e] / (tot + 1e-9f)) * (gacc[64 + e] / (float)NTOK);
    lb *= (float)NEXP;
    float z = gacc[128] / (float)NTOK;
    out[OUT_AUX] = 0.001f * lb + 0.001f * z;
  }
}

// ---------------------------------------------------------------------------
extern "C" void kernel_launch(void* const* d_in, const int* in_sizes, int n_in,
                              void* d_out, int out_size, void* d_ws, size_t ws_size,
                              hipStream_t stream) {
  const float* hidden = (const float*)d_in[0];   // [4,4096,4096] f32
  const float* gate   = (const float*)d_in[1];   // [64,4096]     f32
  const float* bias   = (const float*)d_in[2];   // [64]          f32
  float* out  = (float*)d_out;
  __bf16* gbf = (__bf16*)d_ws;
  float*  gacc = (float*)((char*)d_ws + WS_GATE_BYTES);

  router_prep<<<128, 256, 0, stream>>>(gate, d_ws);
  router_main<<<128, 256, 0, stream>>>(hidden, bias, gbf, out, gacc);
  router_fin<<<1, 64, 0, stream>>>(gacc, out);
}
